// CoarseMatching_70274254897274
// MI455X (gfx1250) — compile-verified, matrix operand order checked by reading, not run on hardware
//
#include <hip/hip_runtime.h>
#include <hip/hip_bf16.h>
#include <math.h>

typedef __attribute__((ext_vector_type(16))) __bf16 v16bf;
typedef __attribute__((ext_vector_type(8)))  __bf16 v8bf;
typedef __attribute__((ext_vector_type(8)))  float  v8f;

#define N_ 4
#define L_ 4800
#define S_ 4800
#define C_ 256

// sim = (f1/sqrt(C)) . (f2/sqrt(C))^T / T  ->  raw_dot * (1/C)/T
__device__ __host__ constexpr float kScale2() { return (1.0f / (float)C_) / 0.1f; }
#define K_THRESH 0.2f

// ---------------------------------------------------------------- convert
__global__ __launch_bounds__(256)
void k_convert(const float* __restrict__ src, __bf16* __restrict__ dst, int n) {
    int i = blockIdx.x * blockDim.x + threadIdx.x;
    if (i < n) dst[i] = (__bf16)src[i];
}

// ---------------------------------------------------------------- GEMM (WMMA bf16)
// One wave per block. Wave tile 64(L) x 64(S) = 4x4 WMMA accumulators.
// K loop: 256 in steps of 32 -> 8 fragment loads feed 16 WMMAs per step
// (~32 FLOP/byte from L2; both bf16 operands are L2-resident, ~20MB total).
__global__ __launch_bounds__(32)
void k_gemm(const __bf16* __restrict__ A, const __bf16* __restrict__ B,
            float* __restrict__ sim) {
    const int n    = blockIdx.z;
    const int lb   = blockIdx.x * 64;        // L tile base
    const int sb   = blockIdx.y * 64;        // S tile base
    const int lane = threadIdx.x & 31;
    const int mrow = lane & 15;
    const int hi   = lane >> 4;

    const __bf16* Abase = A + (size_t)n * L_ * C_;
    const __bf16* Bbase = B + (size_t)n * S_ * C_;

    v8f acc[4][4] = {};

    for (int k = 0; k < C_; k += 32) {
        v16bf afrag[4], bfrag[4];
        #pragma unroll
        for (int i = 0; i < 4; ++i) {
            // ISA 16-bit A 16x32 layout: lane<16 holds K 0..7 & 16..23,
            // lane>=16 holds K 8..15 & 24..31 (two contiguous 16B chunks).
            const __bf16* pa = Abase + (size_t)(lb + 16 * i + mrow) * C_ + k;
            v8bf a0 = *(const v8bf*)(pa + hi * 8);
            v8bf a1 = *(const v8bf*)(pa + 16 + hi * 8);
            afrag[i] = __builtin_shufflevector(a0, a1,
                0,1,2,3,4,5,6,7,8,9,10,11,12,13,14,15);
            // B used as K x N with column n = row n of f2 (A.B^T): symmetric load.
            const __bf16* pb = Bbase + (size_t)(sb + 16 * i + mrow) * C_ + k;
            v8bf b0 = *(const v8bf*)(pb + hi * 8);
            v8bf b1 = *(const v8bf*)(pb + 16 + hi * 8);
            bfrag[i] = __builtin_shufflevector(b0, b1,
                0,1,2,3,4,5,6,7,8,9,10,11,12,13,14,15);
        }
        #pragma unroll
        for (int i = 0; i < 4; ++i)
            #pragma unroll
            for (int j = 0; j < 4; ++j)
                acc[i][j] = __builtin_amdgcn_wmma_f32_16x16x32_bf16(
                    false, afrag[i], false, bfrag[j],
                    (short)0, acc[i][j], false, false);
    }

    // C/D layout: VGPR r, lane -> M = r + 8*(lane>>4), N = lane&15.
    #pragma unroll
    for (int i = 0; i < 4; ++i)
        #pragma unroll
        for (int j = 0; j < 4; ++j)
            #pragma unroll
            for (int r = 0; r < 8; ++r) {
                int m = lb + 16 * i + r + 8 * hi;
                int s = sb + 16 * j + (lane & 15);
                sim[((size_t)n * L_ + m) * S_ + s] = acc[i][j][r] * kScale2();
            }
}

// ---------------------------------------------------------------- row softmax stats
__global__ __launch_bounds__(256)
void k_rowstat(const float* __restrict__ sim,
               float* __restrict__ rowmax, float* __restrict__ rowsum) {
    const int n = blockIdx.y, l = blockIdx.x;
    const float* row = sim + ((size_t)n * L_ + l) * S_;
    float m = -INFINITY, s = 0.f;
    for (int j = threadIdx.x; j < S_; j += 256) {
        float x = row[j];
        if (x > m) { s = s * __expf(m - x) + 1.f; m = x; }
        else       { s += __expf(x - m); }
    }
    __shared__ float sm[256], ss[256];
    sm[threadIdx.x] = m; ss[threadIdx.x] = s;
    __syncthreads();
    for (int off = 128; off; off >>= 1) {
        if ((int)threadIdx.x < off) {
            float m2 = sm[threadIdx.x + off], s2 = ss[threadIdx.x + off];
            float M  = fmaxf(sm[threadIdx.x], m2);
            ss[threadIdx.x] = ss[threadIdx.x] * __expf(sm[threadIdx.x] - M)
                            + s2 * __expf(m2 - M);
            sm[threadIdx.x] = M;
        }
        __syncthreads();
    }
    if (threadIdx.x == 0) { rowmax[n * L_ + l] = sm[0]; rowsum[n * L_ + l] = ss[0]; }
}

// ---------------------------------------------------------------- column softmax stats
__global__ __launch_bounds__(256)
void k_colstat(const float* __restrict__ sim,
               float* __restrict__ colmax, float* __restrict__ colsum) {
    const int n = blockIdx.y;
    const int s = blockIdx.x * 256 + threadIdx.x;
    if (s >= S_) return;
    const float* base = sim + (size_t)n * L_ * S_ + s;
    float m = -INFINITY, acc = 0.f;
    for (int l = 0; l < L_; ++l) {
        if (l + 8 < L_) __builtin_prefetch(base + (size_t)(l + 8) * S_, 0, 1);
        float x = base[(size_t)l * S_];
        if (x > m) { acc = acc * __expf(m - x) + 1.f; m = x; }
        else       { acc += __expf(x - m); }
    }
    colmax[n * S_ + s] = m;
    colsum[n * S_ + s] = acc;
}

// ---------------------------------------------------------------- conf + row conf max
__global__ __launch_bounds__(256)
void k_conf(const float* __restrict__ sim,
            const float* __restrict__ rowmax, const float* __restrict__ rowsum,
            const float* __restrict__ colmax, const float* __restrict__ colsum,
            float* __restrict__ conf_out, float* __restrict__ rowconfmax) {
    const int n = blockIdx.y, l = blockIdx.x;
    const float rm   = rowmax[n * L_ + l];
    const float rinv = 1.f / rowsum[n * L_ + l];
    const float* row  = sim      + ((size_t)n * L_ + l) * S_;
    float*       crow = conf_out + ((size_t)n * L_ + l) * S_;
    float mx = 0.f;
    for (int s = threadIdx.x; s < S_; s += 256) {
        float x = row[s];
        float a = __expf(x - rm) * rinv;                      // softmax over S (axis 2)
        float b = __expf(x - colmax[n * S_ + s]) / colsum[n * S_ + s]; // over L (axis 1)
        float c = a * b;
        crow[s] = c;
        mx = fmaxf(mx, c);
    }
    __shared__ float sm[256];
    sm[threadIdx.x] = mx;
    __syncthreads();
    for (int off = 128; off; off >>= 1) {
        if ((int)threadIdx.x < off)
            sm[threadIdx.x] = fmaxf(sm[threadIdx.x], sm[threadIdx.x + off]);
        __syncthreads();
    }
    if (threadIdx.x == 0) rowconfmax[n * L_ + l] = sm[0];
}

// ---------------------------------------------------------------- per-column conf max
__global__ __launch_bounds__(256)
void k_colconfmax(const float* __restrict__ conf, float* __restrict__ colconfmax) {
    const int n = blockIdx.y;
    const int s = blockIdx.x * 256 + threadIdx.x;
    if (s >= S_) return;
    const float* base = conf + (size_t)n * L_ * S_ + s;
    float m = 0.f;   // conf >= 0 always
    for (int l = 0; l < L_; ++l) {
        if (l + 8 < L_) __builtin_prefetch(base + (size_t)(l + 8) * S_, 0, 1);
        m = fmaxf(m, base[(size_t)l * S_]);
    }
    colconfmax[n * S_ + s] = m;
}

// ---------------------------------------------------------------- mask + extraction
__global__ __launch_bounds__(256)
void k_final(const float* __restrict__ conf,
             const float* __restrict__ rowconfmax, const float* __restrict__ colconfmax,
             float* __restrict__ mask_out, float* __restrict__ matchconf,
             float* __restrict__ validv, float* __restrict__ colidx) {
    const int n = blockIdx.y, l = blockIdx.x;
    const float rcm = rowconfmax[n * L_ + l];
    const float* crow = conf     + ((size_t)n * L_ + l) * S_;
    float*       mrow = mask_out + ((size_t)n * L_ + l) * S_;
    int best = 0x7fffffff;
    for (int s = threadIdx.x; s < S_; s += 256) {
        float c = crow[s];
        bool hit = (c > K_THRESH) && (c == rcm) && (c == colconfmax[n * S_ + s]);
        mrow[s] = hit ? 1.f : 0.f;
        if (hit && s < best) best = s;
    }
    __shared__ int si[256];
    si[threadIdx.x] = best;
    __syncthreads();
    for (int off = 128; off; off >>= 1) {
        if ((int)threadIdx.x < off)
            si[threadIdx.x] = min(si[threadIdx.x], si[threadIdx.x + off]);
        __syncthreads();
    }
    if (threadIdx.x == 0) {
        int  b = si[0];
        bool v = (b != 0x7fffffff);          // jnp.any
        int  idx = v ? b : 0;                // jnp.argmax of bool mask = first True, else 0
        validv[n * L_ + l]    = v ? 1.f : 0.f;
        colidx[n * L_ + l]    = (float)idx;
        matchconf[n * L_ + l] = v ? crow[idx] : 0.f;
    }
}

// ---------------------------------------------------------------- launcher
extern "C" void kernel_launch(void* const* d_in, const int* in_sizes, int n_in,
                              void* d_out, int out_size, void* d_ws, size_t ws_size,
                              hipStream_t stream) {
    (void)in_sizes; (void)n_in; (void)out_size; (void)ws_size;

    const float* f1 = (const float*)d_in[0];   // [N,L,C] f32
    const float* f2 = (const float*)d_in[1];   // [N,S,C] f32

    const size_t NL  = (size_t)N_ * L_;
    const size_t NS  = (size_t)N_ * S_;
    const size_t NLC = (size_t)N_ * L_ * C_;
    const size_t NLS = (size_t)N_ * L_ * S_;

    // d_out layout (flat, return order, all f32):
    float* out_matchconf = (float*)d_out;                 // [N*L]
    float* out_valid     = out_matchconf + NL;            // [N*L]
    float* out_colidx    = out_valid + NL;                // [N*L]
    float* out_mask      = out_colidx + NL;               // [N*L*S]
    float* out_conf      = out_mask + NLS;                // [N*L*S]

    // The mask slot doubles as sim scratch (368MB) until k_final overwrites it.
    float* sim = out_mask;

    // Workspace (small, ~20MB): bf16 operands + softmax stats.
    char* ws = (char*)d_ws;
    size_t off = 0;
    auto take = [&](size_t bytes) { char* p = ws + off; off = (off + bytes + 255) & ~(size_t)255; return p; };
    __bf16* f1b       = (__bf16*)take(NLC * sizeof(__bf16));
    __bf16* f2b       = (__bf16*)take(NLC * sizeof(__bf16));
    float*  rowmax    = (float*)take(NL * sizeof(float));
    float*  rowsum    = (float*)take(NL * sizeof(float));
    float*  colmax    = (float*)take(NS * sizeof(float));
    float*  colsum    = (float*)take(NS * sizeof(float));
    float*  rowconfmx = (float*)take(NL * sizeof(float));
    float*  colconfmx = (float*)take(NS * sizeof(float));

    // 1) fp32 -> bf16
    {
        int n = (int)NLC;
        int g = (n + 255) / 256;
        k_convert<<<g, 256, 0, stream>>>(f1, f1b, n);
        k_convert<<<g, 256, 0, stream>>>(f2, f2b, n);
    }
    // 2) WMMA GEMM: sim[n,l,s]
    {
        dim3 grid(L_ / 64, S_ / 64, N_);   // 75 x 75 x 4, one wave per block
        k_gemm<<<grid, 32, 0, stream>>>(f1b, f2b, sim);
    }
    // 3) softmax stats
    k_rowstat<<<dim3(L_, N_), 256, 0, stream>>>(sim, rowmax, rowsum);
    k_colstat<<<dim3((S_ + 255) / 256, N_), 256, 0, stream>>>(sim, colmax, colsum);
    // 4) conf + per-row conf max
    k_conf<<<dim3(L_, N_), 256, 0, stream>>>(sim, rowmax, rowsum, colmax, colsum,
                                             out_conf, rowconfmx);
    // 5) per-column conf max
    k_colconfmax<<<dim3((S_ + 255) / 256, N_), 256, 0, stream>>>(out_conf, colconfmx);
    // 6) mask (overwrites sim scratch) + per-row extraction
    k_final<<<dim3(L_, N_), 256, 0, stream>>>(out_conf, rowconfmx, colconfmx,
                                              out_mask, out_matchconf, out_valid, out_colidx);
}
